// SkipGPTAttention_5970004541937
// MI455X (gfx1250) — matrix operand order repaired and checked by reading, not compiled, not run on hardware
//
#include <hip/hip_runtime.h>
#include <hip/hip_bf16.h>

typedef __bf16 bf16_t;
typedef __attribute__((ext_vector_type(16))) __bf16 v16bf;
typedef __attribute__((ext_vector_type(8)))  __bf16 v8bf;
typedef __attribute__((ext_vector_type(8)))  float  v8f;
typedef __attribute__((ext_vector_type(4)))  float  v4f;
typedef __attribute__((ext_vector_type(4)))  unsigned int v4u;
typedef __attribute__((ext_vector_type(8)))  int    v8i_;
typedef __attribute__((ext_vector_type(4)))  int    v4i_;

#define B_  2
#define L_  2048
#define D_  4096
#define HQ_ 32
#define HK_ 8
#define DH_ 128
#define G_  (HQ_ / HK_)
#define SCALING_ 0.08838834764831845f   // 128^-0.5

#if defined(__has_builtin)
#if __has_builtin(__builtin_amdgcn_tensor_load_to_lds) && \
    __has_builtin(__builtin_amdgcn_s_wait_tensorcnt)
#define USE_TDM 1
#endif
#endif
#ifndef USE_TDM
#define USE_TDM 0
#endif

__device__ inline v8f zero8() {
    v8f z;
#pragma unroll
    for (int i = 0; i < 8; ++i) z[i] = 0.0f;
    return z;
}

__device__ inline v16bf cat16(v8bf lo, v8bf hi8) {
    v16bf r;
#pragma unroll
    for (int e = 0; e < 8; ++e) { r[e] = lo[e]; r[e + 8] = hi8[e]; }
    return r;
}

#if USE_TDM
// 2D TDM load: 128x128 bf16 tile from tensor (dim1 x dim0, row stride stride0
// elements) starting at gaddr, into LDS byte offset ldsOff.
__device__ inline void tdm_load_2d(unsigned ldsOff, const bf16_t* gaddr,
                                   int dim0, int dim1, int stride0) {
    unsigned long long ga = (unsigned long long)(const void*)gaddr;
    v4u g0;
    g0[0] = 1u;                                    // count=1, user descriptor
    g0[1] = ldsOff;                                // lds_addr (bytes)
    g0[2] = (unsigned)(ga & 0xffffffffu);          // global_addr[31:0]
    g0[3] = (unsigned)((ga >> 32) & 0x01ffffffu)   // global_addr[56:32]
            | 0x80000000u;                         // type = 2 ("image")
    v8i_ g1;
    g1[0] = 0x00010000;                            // wg_mask=0, data_size=1 (2B)
    g1[1] = (dim0 & 0xffff) << 16;                 // tensor_dim0 low half
    g1[2] = ((dim0 >> 16) & 0xffff) | ((dim1 & 0xffff) << 16);
    g1[3] = ((dim1 >> 16) & 0xffff) | (128 << 16); // tile_dim0 = 128
    g1[4] = 128;                                   // tile_dim1 = 128, tile_dim2 = 0
    g1[5] = stride0;                               // tensor_dim0_stride (elements)
    g1[6] = 0;
    g1[7] = 0;
    v4i_ gz = {0, 0, 0, 0};
#if defined(__clang_major__) && (__clang_major__ >= 23)
    v8i_ gz8 = {0, 0, 0, 0, 0, 0, 0, 0};
    __builtin_amdgcn_tensor_load_to_lds(g0, g1, gz, gz, gz8, 0);
#else
    __builtin_amdgcn_tensor_load_to_lds(g0, g1, gz, gz, 0);
#endif
}
#endif

// ---------------------------------------------------------------------------
// Generic bf16-WMMA GEMM: C[m,n] = sum_k A[m,k] * Bw[n,k], optional row mask.
// A: (M,K) fp32 row-major. Bw: (N,K) fp32 row-major. C: (M,N) fp32.
// M,N multiples of 128; K multiple of 32. Block = 128x128 tile, 256 threads.
// ---------------------------------------------------------------------------
__global__ __launch_bounds__(256) void gemm_rowmask_kernel(
    const float* __restrict__ A, const float* __restrict__ Bw,
    float* __restrict__ C, int M, int N, int K,
    const int* __restrict__ rowmask)
{
    __shared__ bf16_t As[128 * 40];   // row stride 40 bf16 (80B, 16B-aligned groups)
    __shared__ bf16_t Bs[128 * 40];

    const int tid     = threadIdx.x;
    const int lane    = tid & 31;
    const int wave    = tid >> 5;
    const int wm      = wave >> 1;        // 0..3 -> 32-row strip
    const int wn      = wave & 1;         // 0..1 -> 64-col strip
    const int mBase   = blockIdx.y * 128;
    const int nBase   = blockIdx.x * 128;
    const int hiHalf  = lane >> 4;        // 0 or 1
    const int l16     = lane & 15;

    v8f acc[2][4];
#pragma unroll
    for (int i = 0; i < 2; ++i)
#pragma unroll
        for (int j = 0; j < 4; ++j) acc[i][j] = zero8();

    for (int kt = 0; kt < K; kt += 32) {
        // Cooperative load of 128x32 fp32 tiles of A and Bw, convert to bf16.
#pragma unroll
        for (int it = 0; it < 4; ++it) {
            int idx = (tid + it * 256) * 4;      // 4096 floats per tile
            int r = idx >> 5, c = idx & 31;
            v4f av = *(const v4f*)(A  + (size_t)(mBase + r) * K + kt + c);
            v4f bv = *(const v4f*)(Bw + (size_t)(nBase + r) * K + kt + c);
            bf16_t* ap = &As[r * 40 + c];
            bf16_t* bp = &Bs[r * 40 + c];
#pragma unroll
            for (int e = 0; e < 4; ++e) { ap[e] = (bf16_t)av[e]; bp[e] = (bf16_t)bv[e]; }
        }
        // Prefetch next k-step tiles into cache (global_prefetch_b8).
        if (kt + 32 < K) {
            __builtin_prefetch(A  + (size_t)(mBase + (tid >> 1)) * K + kt + 32, 0, 1);
            __builtin_prefetch(Bw + (size_t)(nBase + (tid >> 1)) * K + kt + 32, 0, 1);
        }
        __syncthreads();

        // A-operand frags (16x32): lane<16 holds K 0..7 & 16..23 of row l16.
        v16bf afrag[2];
#pragma unroll
        for (int i = 0; i < 2; ++i) {
            int r = wm * 32 + i * 16 + l16;
            v8bf lo = *(const v8bf*)&As[r * 40 + hiHalf * 8];
            v8bf hb = *(const v8bf*)&As[r * 40 + hiHalf * 8 + 16];
            afrag[i] = cat16(lo, hb);
        }
        // B-operand frags (32x16): lane<16 holds K 0..7 & 16..23 of col l16.
        v16bf bfrag[4];
#pragma unroll
        for (int j = 0; j < 4; ++j) {
            int r = wn * 64 + j * 16 + l16;
            v8bf lo = *(const v8bf*)&Bs[r * 40 + hiHalf * 8];
            v8bf hb = *(const v8bf*)&Bs[r * 40 + hiHalf * 8 + 16];
            bfrag[j] = cat16(lo, hb);
        }
#pragma unroll
        for (int i = 0; i < 2; ++i)
#pragma unroll
            for (int j = 0; j < 4; ++j)
                acc[i][j] = __builtin_amdgcn_wmma_f32_16x16x32_bf16(
                    false, afrag[i], false, bfrag[j], (short)0, acc[i][j], false, false);
        __syncthreads();
    }

    // Row-mask values hoisted out of the store loops (uniform branch, 16 loads).
    float mv[2][8];
    if (rowmask) {
#pragma unroll
        for (int i = 0; i < 2; ++i)
#pragma unroll
            for (int r = 0; r < 8; ++r)
                mv[i][r] = (float)rowmask[mBase + wm * 32 + i * 16 + hiHalf * 8 + r];
    } else {
#pragma unroll
        for (int i = 0; i < 2; ++i)
#pragma unroll
            for (int r = 0; r < 8; ++r) mv[i][r] = 1.0f;
    }

    // Store (C layout: vgpr r -> row r + 8*hiHalf; col = l16)
#pragma unroll
    for (int i = 0; i < 2; ++i) {
#pragma unroll
        for (int j = 0; j < 4; ++j) {
            int col = nBase + wn * 64 + j * 16 + l16;
#pragma unroll
            for (int r = 0; r < 8; ++r) {
                int m = mBase + wm * 32 + i * 16 + hiHalf * 8 + r;
                C[(size_t)m * N + col] = acc[i][j][r] * mv[i][r];
            }
        }
    }
}

// ---------------------------------------------------------------------------
// RoPE + layout transform: (B*L, H*DH) fp32 -> (B, H, L, DH) bf16 (Q and K).
// ---------------------------------------------------------------------------
__global__ __launch_bounds__(256) void rope_convert_kernel(
    const float* __restrict__ in, const float* __restrict__ cosb,
    const float* __restrict__ sinb, bf16_t* __restrict__ out, int nheads)
{
    int idx = blockIdx.x * 256 + threadIdx.x;      // over (b, h, l, dh)
    int dh = idx & (DH_ - 1);
    int l  = (idx >> 7) & (L_ - 1);
    int bh = idx >> 18;                            // L_*DH_ = 2^18
    int h  = bh % nheads;
    int b  = bh / nheads;
    size_t inIdx = (((size_t)(b * L_ + l)) * nheads + h) * DH_ + dh;
    size_t cs = ((size_t)(b * L_ + l)) * DH_ + dh;
    float v = in[inIdx];
    float c = cosb[cs];
    float s = sinb[cs];
    float partner = in[inIdx + (dh < 64 ? 64 : -64)];
    float rot = (dh < 64) ? -partner : partner;
    out[idx] = (bf16_t)(v * c + rot * s);
}

// ---------------------------------------------------------------------------
// V transpose + convert: (B, L, HK, DH) fp32 -> (B, HK, DH, L) bf16.
// 32x32 LDS tile, coalesced on both sides; done ONCE per element (the flash
// kernel previously re-transposed each V tile up to 64x).
// ---------------------------------------------------------------------------
__global__ __launch_bounds__(256) void transpose_convert_kernel(
    const float* __restrict__ in, bf16_t* __restrict__ out)
{
    __shared__ float tile[32][33];
    const int bh = blockIdx.z;                 // b*HK + h
    const int b = bh / HK_, h = bh % HK_;
    const int l0 = blockIdx.x * 32;
    const int d0 = blockIdx.y * 32;
    const int tx = threadIdx.x & 31;
    const int ty = threadIdx.x >> 5;           // 0..7
#pragma unroll
    for (int i = 0; i < 4; ++i) {
        int l = l0 + ty + i * 8;
        tile[ty + i * 8][tx] =
            in[(((size_t)(b * L_ + l)) * HK_ + h) * DH_ + d0 + tx];
    }
    __syncthreads();
#pragma unroll
    for (int i = 0; i < 4; ++i) {
        int d = d0 + ty + i * 8;
        out[(((size_t)(b * HK_ + h)) * DH_ + d) * L_ + l0 + tx] =
            (bf16_t)tile[tx][ty + i * 8];
    }
}

// ---------------------------------------------------------------------------
// Flash attention (causal, GQA). Grid: (L/128, B*HQ). 256 threads = 8 waves.
// K tile (key,dh) and V tile (dh,key) are both fetched by the Tensor Data
// Mover, double-buffered so the DMA overlaps WMMA compute. LDS map (bytes):
//   [0..64K)    K tiles  x2 (128x128 bf16 each)
//   [64K..128K) V tiles  x2 (128x128 bf16 each, dh-major)
//   [128K..160K) P scratch, 8 waves x (16x128 bf16)
// Q: (B,HQ,L,DH) bf16; K: (B,HK,L,DH) bf16; Vt: (B,HK,DH,L) bf16;
// Out: (B,L,HQ*DH) fp32.
// ---------------------------------------------------------------------------
__global__ __launch_bounds__(256) void flash_attn_kernel(
    const bf16_t* __restrict__ Qb, const bf16_t* __restrict__ Kb,
    const bf16_t* __restrict__ Vt, float* __restrict__ Out)
{
    extern __shared__ char smem[];

    const int tid    = threadIdx.x;
    const int lane   = tid & 31;
    const int wave   = tid >> 5;
    const int hiHalf = lane >> 4;
    const int l16    = lane & 15;
    const int qblk   = blockIdx.x;
    const int bh     = blockIdx.y;
    const int hq     = bh % HQ_;
    const int b      = bh / HQ_;
    const int hk     = hq / G_;

    bf16_t* Pw = (bf16_t*)(smem + 131072) + wave * 16 * 128;

    // Q fragments kept in registers: A-operand row = l16 for both lane halves.
    const int qrowA = qblk * 128 + wave * 16 + l16;
    const bf16_t* qbase = Qb + (((size_t)(b * HQ_ + hq)) * L_ + qrowA) * DH_;
    v16bf qf[4];
#pragma unroll
    for (int c = 0; c < 4; ++c) {
        v8bf lo = *(const v8bf*)(qbase + c * 32 + hiHalf * 8);
        v8bf hb = *(const v8bf*)(qbase + c * 32 + hiHalf * 8 + 16);
        qf[c] = cat16(lo, hb);
    }

    v8f oacc[8];
#pragma unroll
    for (int c16 = 0; c16 < 8; ++c16) oacc[c16] = zero8();
    float mrow[8], lrow[8];
#pragma unroll
    for (int r = 0; r < 8; ++r) { mrow[r] = -1e30f; lrow[r] = 0.0f; }

    const bf16_t* kb = Kb + ((size_t)(b * HK_ + hk)) * L_ * DH_;   // (key, dh)
    const bf16_t* vt = Vt + ((size_t)(b * HK_ + hk)) * DH_ * L_;   // (dh, key)

#if USE_TDM
    // Prologue: DMA block 0 into buffer 0.
    if (wave == 0) {
        tdm_load_2d(0u, kb, DH_, L_, DH_);                 // K: tensor (L x DH)
        tdm_load_2d(65536u, vt, L_, DH_, L_);              // V: tensor (DH x L)
    }
#endif

    for (int kblk = 0; kblk <= qblk; ++kblk) {
        const int buf = kblk & 1;
        bf16_t* Ksb = (bf16_t*)(smem + buf * 32768);
        bf16_t* Vsb = (bf16_t*)(smem + 65536 + buf * 32768);

#if USE_TDM
        if (wave == 0) __builtin_amdgcn_s_wait_tensorcnt(0);
        __syncthreads();   // data ready + all waves done with buf^1
        if (wave == 0 && kblk < qblk) {
            // DMA next block into the other buffer; overlaps with compute below.
            tdm_load_2d((unsigned)((buf ^ 1) * 32768),
                        kb + (size_t)((kblk + 1) * 128) * DH_, DH_, L_, DH_);
            tdm_load_2d((unsigned)(65536 + (buf ^ 1) * 32768),
                        vt + (size_t)((kblk + 1) * 128), L_, DH_, L_);
        }
#else
        __syncthreads();
        // Cooperative fallback: K rows (key,dh) and V rows (dh,key).
#pragma unroll
        for (int it = 0; it < 8; ++it) {
            int idx = (tid + it * 256) * 8;      // 16384 bf16 per tile
            int r = idx >> 7, c = idx & 127;
            *(v8bf*)(Ksb + r * 128 + c) =
                *(const v8bf*)(kb + ((size_t)(kblk * 128 + r)) * DH_ + c);
            *(v8bf*)(Vsb + r * 128 + c) =
                *(const v8bf*)(vt + (size_t)r * L_ + kblk * 128 + c);
        }
        __syncthreads();
#endif

        // S = Q * K^T  (8 tiles of 16 keys; K-dim = DH in 4 chunks of 32)
        v8f s[8];
#pragma unroll
        for (int t = 0; t < 8; ++t) {
            v8f a = zero8();
#pragma unroll
            for (int c = 0; c < 4; ++c) {
                const bf16_t* kp = Ksb + (t * 16 + l16) * 128 + c * 32 + hiHalf * 8;
                v8bf lo = *(const v8bf*)kp;
                v8bf hb = *(const v8bf*)(kp + 16);
                a = __builtin_amdgcn_wmma_f32_16x16x32_bf16(
                        false, qf[c], false, cat16(lo, hb), (short)0, a, false, false);
            }
            s[t] = a;
        }

        // Scale + causal mask (only the diagonal block needs masking).
        const int qrow0 = qblk * 128 + wave * 16 + hiHalf * 8;
#pragma unroll
        for (int t = 0; t < 8; ++t) {
            int key = kblk * 128 + t * 16 + l16;
#pragma unroll
            for (int r = 0; r < 8; ++r) {
                float v = s[t][r] * SCALING_;
                if (kblk == qblk && key > qrow0 + r) v = -1e30f;
                s[t][r] = v;
            }
        }

        // Online softmax: per-row max/sum via 16-lane xor-shuffle reductions.
        float mnew[8], fac[8];
#pragma unroll
        for (int r = 0; r < 8; ++r) {
            float mx = mrow[r];
#pragma unroll
            for (int t = 0; t < 8; ++t) mx = fmaxf(mx, s[t][r]);
#pragma unroll
            for (int off = 8; off >= 1; off >>= 1)
                mx = fmaxf(mx, __shfl_xor(mx, off, 32));
            mnew[r] = mx;
            fac[r]  = __expf(mrow[r] - mx);
            mrow[r] = mx;
        }
#pragma unroll
        for (int r = 0; r < 8; ++r) {
            float sum = 0.0f;
#pragma unroll
            for (int t = 0; t < 8; ++t) {
                float p = __expf(s[t][r] - mnew[r]);
                sum += p;
                Pw[(r + hiHalf * 8) * 128 + t * 16 + l16] = (bf16_t)p;
            }
#pragma unroll
            for (int off = 8; off >= 1; off >>= 1)
                sum += __shfl_xor(sum, off, 32);
            lrow[r] = lrow[r] * fac[r] + sum;
        }
        // Rescale O accumulators (same C-layout row mapping as S).
#pragma unroll
        for (int c16 = 0; c16 < 8; ++c16)
#pragma unroll
            for (int r = 0; r < 8; ++r) oacc[c16][r] *= fac[r];

        // O += P * V  (K-dim = 128 keys in 4 chunks of 32)
#pragma unroll
        for (int ks = 0; ks < 4; ++ks) {
            const bf16_t* pp = Pw + l16 * 128 + ks * 32 + hiHalf * 8;
            v8bf plo = *(const v8bf*)pp;
            v8bf phb = *(const v8bf*)(pp + 16);
            v16bf pf = cat16(plo, phb);
#pragma unroll
            for (int c16 = 0; c16 < 8; ++c16) {
                const bf16_t* vp = Vsb + (c16 * 16 + l16) * 128 + ks * 32 + hiHalf * 8;
                v8bf vlo = *(const v8bf*)vp;
                v8bf vhb = *(const v8bf*)(vp + 16);
                oacc[c16] = __builtin_amdgcn_wmma_f32_16x16x32_bf16(
                                false, pf, false, cat16(vlo, vhb), (short)0,
                                oacc[c16], false, false);
            }
        }
    }

    // Epilogue: multiply by reciprocal row sums, write (B, L, HQ*DH) fp32.
    float rinv[8];
#pragma unroll
    for (int r = 0; r < 8; ++r) rinv[r] = 1.0f / lrow[r];
    const int ql0 = qblk * 128 + wave * 16 + hiHalf * 8;
#pragma unroll
    for (int c16 = 0; c16 < 8; ++c16) {
#pragma unroll
        for (int r = 0; r < 8; ++r) {
            int ql = ql0 + r;
            Out[(((size_t)(b * L_ + ql)) * HQ_ + hq) * DH_ + c16 * 16 + l16] =
                oacc[c16][r] * rinv[r];
        }
    }
}

// ---------------------------------------------------------------------------
extern "C" void kernel_launch(void* const* d_in, const int* in_sizes, int n_in,
                              void* d_out, int out_size, void* d_ws, size_t ws_size,
                              hipStream_t stream) {
    (void)in_sizes; (void)n_in; (void)out_size; (void)ws_size;
    const float* hidden = (const float*)d_in[0];
    const float* cosb   = (const float*)d_in[1];
    const float* sinb   = (const float*)d_in[2];
    const float* wq     = (const float*)d_in[3];
    const float* wk     = (const float*)d_in[4];
    const float* wv     = (const float*)d_in[5];
    const float* wo     = (const float*)d_in[6];
    const int*   route  = (const int*)d_in[7];
    float* out = (float*)d_out;

    const size_t SZ_QRAW = (size_t)B_ * L_ * HQ_ * DH_ * sizeof(float);  // 64 MB
    const size_t SZ_KRAW = (size_t)B_ * L_ * HK_ * DH_ * sizeof(float);  // 16 MB

    char* p = (char*)d_ws;
    float*  Qraw = (float*)p;  p += SZ_QRAW;      // reused as attn output later
    float*  Kraw = (float*)p;  p += SZ_KRAW;
    float*  Vraw = (float*)p;  p += SZ_KRAW;
    bf16_t* Qb   = (bf16_t*)p; p += SZ_QRAW / 2;
    bf16_t* Kb   = (bf16_t*)p; p += SZ_KRAW / 2;
    bf16_t* Vt   = (bf16_t*)p;                    // (B, HK, DH, L) bf16

    const int M = B_ * L_;           // 4096
    dim3 blk(256);

    // QKV projections (mask folded into Q rows; mask^2 == mask).
    gemm_rowmask_kernel<<<dim3((HQ_ * DH_) / 128, M / 128), blk, 0, stream>>>(
        hidden, wq, Qraw, M, HQ_ * DH_, D_, route);
    gemm_rowmask_kernel<<<dim3((HK_ * DH_) / 128, M / 128), blk, 0, stream>>>(
        hidden, wk, Kraw, M, HK_ * DH_, D_, nullptr);
    gemm_rowmask_kernel<<<dim3((HK_ * DH_) / 128, M / 128), blk, 0, stream>>>(
        hidden, wv, Vraw, M, HK_ * DH_, D_, nullptr);

    // RoPE + (B,H,L,DH) bf16 layout for Q/K; transpose-convert for V.
    rope_convert_kernel<<<(B_ * HQ_ * L_ * DH_) / 256, blk, 0, stream>>>(
        Qraw, cosb, sinb, Qb, HQ_);
    rope_convert_kernel<<<(B_ * HK_ * L_ * DH_) / 256, blk, 0, stream>>>(
        Kraw, cosb, sinb, Kb, HK_);
    transpose_convert_kernel<<<dim3(L_ / 32, DH_ / 32, B_ * HK_), blk, 0, stream>>>(
        Vraw, Vt);

    // Flash attention -> Qraw reused as (B, L, HQ*DH) fp32 attn output.
    size_t smemBytes = 163840;   // 2x32KB K + 2x32KB V + 32KB P
    flash_attn_kernel<<<dim3(L_ / 128, B_ * HQ_), blk, smemBytes, stream>>>(
        Qb, Kb, Vt, Qraw);

    // Output projection with row mask.
    gemm_rowmask_kernel<<<dim3(D_ / 128, M / 128), blk, 0, stream>>>(
        Qraw, wo, out, M, D_, HQ_ * DH_, route);
}
